// NAND_78589311582948
// MI455X (gfx1250) — matrix operand order repaired and checked by reading, not compiled, not run on hardware
//
#include <hip/hip_runtime.h>
#include <math.h>

// Problem constants (from the reference)
#define BATCH    1024
#define NCHOICE  4096   // N_CHOICES * REDUNDANCY
#define NVEC     8
#define VDIM     256
#define ROW      (NVEC * VDIM)   // 2048

typedef __attribute__((ext_vector_type(16))) __bf16 v16bf;
typedef __attribute__((ext_vector_type(8)))  float  v8f;

// ---------------------------------------------------------------------------
// Pre-pass 1: per-row L2 normalization of a 256-float vector, split into
// bf16 hi/lo planes (hi = bf16(x), lo = bf16(x - float(hi))).
// One wave (32 lanes) per row; each lane owns 8 consecutive floats.
// ---------------------------------------------------------------------------
__global__ __launch_bounds__(256) void NAND_norm_split_kernel(
    const float* __restrict__ src,
    __bf16* __restrict__ hi, __bf16* __restrict__ lo, int nrows)
{
    int wid  = (blockIdx.x * blockDim.x + threadIdx.x) >> 5;
    int lane = threadIdx.x & 31;
    if (wid >= nrows) return;

    const float* r = src + (size_t)wid * VDIM;
    float4 x0 = *(const float4*)(r + lane * 8);
    float4 x1 = *(const float4*)(r + lane * 8 + 4);

    float ss = x0.x*x0.x + x0.y*x0.y + x0.z*x0.z + x0.w*x0.w
             + x1.x*x1.x + x1.y*x1.y + x1.z*x1.z + x1.w*x1.w;
#pragma unroll
    for (int off = 16; off > 0; off >>= 1)
        ss += __shfl_xor(ss, off, 32);

    float inv = 1.0f / fmaxf(sqrtf(ss), 1e-8f);   // matches eps clamp in ref

    float v[8] = {x0.x, x0.y, x0.z, x0.w, x1.x, x1.y, x1.z, x1.w};
    __attribute__((aligned(16))) __bf16 hb[8];
    __attribute__((aligned(16))) __bf16 lb[8];
#pragma unroll
    for (int i = 0; i < 8; ++i) {
        float s = v[i] * inv;
        __bf16 h = (__bf16)s;
        hb[i] = h;
        lb[i] = (__bf16)(s - (float)h);
    }
    size_t o = (size_t)wid * VDIM + lane * 8;
    *(uint4*)(hi + o) = *(const uint4*)hb;
    *(uint4*)(lo + o) = *(const uint4*)lb;
}

// ---------------------------------------------------------------------------
// Pre-pass 2: sigmoid coefficients, transposed to [n][C] for per-lane loads.
// interpolated = nw*cos + (1-nw)*(1-cos) = (1-nw) + (2*nw-1)*cos
// ---------------------------------------------------------------------------
__global__ __launch_bounds__(256) void NAND_coeff_kernel(
    const float* __restrict__ nand_weight,
    float* __restrict__ ca, float* __restrict__ cb)
{
    int t = blockIdx.x * blockDim.x + threadIdx.x;
    if (t >= NCHOICE * NVEC) return;
    int c = t >> 3;       // row in [C, NVEC]
    int n = t & 7;
    float s = 1.0f / (1.0f + expf(-nand_weight[t]));
    ca[n * NCHOICE + c] = 2.0f * s - 1.0f;
    cb[n * NCHOICE + c] = 1.0f - s;
}

// ---------------------------------------------------------------------------
// Fragment loaders (layouts per CDNA5 ISA 7.12.2, wave32):
//
// A (16x32 bf16, M x K): lane L<16 -> row M=L, holds K = [0..7] in v0..v3 and
// K = [16..23] in v4..v7; lane L>=16 -> row M=L-16, K = [8..15] and [24..31].
// => per lane: two contiguous 8-element (16B) chunks at K = half*8 and 16+half*8.
//
// B (32x16 bf16, K x N): same K order as A, lane = column N. Since B[k][n] = wn[c=n][k],
// lane n holds two 8-element chunks of weight row n at K = half*8 and 16+half*8.
// ---------------------------------------------------------------------------
__device__ __forceinline__ v16bf NAND_load2x8(const __bf16* p0, const __bf16* p1)
{
    union { v16bf v; uint4 u[2]; } r;
    r.u[0] = *(const uint4*)p0;
    r.u[1] = *(const uint4*)p1;
    return r.v;
}

__device__ __forceinline__ v8f NAND_wmma3(v16bf ah, v16bf al, v16bf bh, v16bf bl, v8f acc)
{
    acc = __builtin_amdgcn_wmma_f32_16x16x32_bf16(false, ah, false, bh, (short)0, acc, false, false);
    acc = __builtin_amdgcn_wmma_f32_16x16x32_bf16(false, ah, false, bl, (short)0, acc, false, false);
    acc = __builtin_amdgcn_wmma_f32_16x16x32_bf16(false, al, false, bh, (short)0, acc, false, false);
    return acc;
}

// ---------------------------------------------------------------------------
// Main kernel: one wave per 32x32 (b,c) output tile (2x2 register blocking of
// 16x16 WMMA tiles) -> 12 WMMAs per 16 b128 loads per 32-K chunk.
// cos = qhi.whi + qhi.wlo + qlo.whi  (split-precision bf16 WMMA, f32 accum)
// ---------------------------------------------------------------------------
__global__ __launch_bounds__(256) void NAND_wmma_kernel(
    const __bf16* __restrict__ qhi, const __bf16* __restrict__ qlo,
    const __bf16* __restrict__ whi, const __bf16* __restrict__ wlo,
    const float*  __restrict__ ca,  const float*  __restrict__ cb,
    float* __restrict__ out)
{
    const int CT = NCHOICE / 32;                       // 128 column supertiles
    int wid  = (blockIdx.x * blockDim.x + threadIdx.x) >> 5;
    int lane = threadIdx.x & 31;
    int btile = wid / CT;                              // 32-row batch supertile
    int ctile = wid % CT;                              // 32-col choice supertile
    int half  = lane >> 4;
    int l16   = lane & 15;

    // A fragment row pointers (two 16-row subtiles), B likewise.
    const __bf16* qh0 = qhi + (size_t)(btile * 32 + l16) * ROW;
    const __bf16* qh1 = qh0 + (size_t)16 * ROW;
    const __bf16* ql0 = qlo + (size_t)(btile * 32 + l16) * ROW;
    const __bf16* ql1 = ql0 + (size_t)16 * ROW;
    const __bf16* wh0 = whi + (size_t)(ctile * 32 + l16) * ROW;
    const __bf16* wh1 = wh0 + (size_t)16 * ROW;
    const __bf16* wl0 = wlo + (size_t)(ctile * 32 + l16) * ROW;
    const __bf16* wl1 = wl0 + (size_t)16 * ROW;
    int crow0 = ctile * 32 + l16;                      // column for N-subtile 0
    int crow1 = crow0 + 16;                            // column for N-subtile 1

    v8f prod00, prod01, prod10, prod11;
#pragma unroll
    for (int i = 0; i < 8; ++i) {
        prod00[i] = 1.0f; prod01[i] = 1.0f;
        prod10[i] = 1.0f; prod11[i] = 1.0f;
    }

    for (int n = 0; n < NVEC; ++n) {
        v8f acc00 = {}, acc01 = {}, acc10 = {}, acc11 = {};
#pragma unroll
        for (int kc = 0; kc < VDIM / 32; ++kc) {
            int kb = n * VDIM + kc * 32;
            int ao = kb + half * 8;                    // A-fragment chunk base
            int bo = kb + half * 8;                    // B-fragment chunk base
            v16bf ah0 = NAND_load2x8(qh0 + ao, qh0 + ao + 16);
            v16bf al0 = NAND_load2x8(ql0 + ao, ql0 + ao + 16);
            v16bf ah1 = NAND_load2x8(qh1 + ao, qh1 + ao + 16);
            v16bf al1 = NAND_load2x8(ql1 + ao, ql1 + ao + 16);
            v16bf bh0 = NAND_load2x8(wh0 + bo, wh0 + bo + 16);
            v16bf bl0 = NAND_load2x8(wl0 + bo, wl0 + bo + 16);
            v16bf bh1 = NAND_load2x8(wh1 + bo, wh1 + bo + 16);
            v16bf bl1 = NAND_load2x8(wl1 + bo, wl1 + bo + 16);
            acc00 = NAND_wmma3(ah0, al0, bh0, bl0, acc00);
            acc01 = NAND_wmma3(ah0, al0, bh1, bl1, acc01);
            acc10 = NAND_wmma3(ah1, al1, bh0, bl0, acc10);
            acc11 = NAND_wmma3(ah1, al1, bh1, bl1, acc11);
        }
        // C/D layout: VGPR i -> (M = i + half*8, N = l16); coefficient depends
        // only on the column c and n -> per-lane scalars.
        float a0 = ca[n * NCHOICE + crow0], b0 = cb[n * NCHOICE + crow0];
        float a1 = ca[n * NCHOICE + crow1], b1 = cb[n * NCHOICE + crow1];
#pragma unroll
        for (int i = 0; i < 8; ++i) {
            prod00[i] *= fmaf(a0, acc00[i], b0);
            prod01[i] *= fmaf(a1, acc01[i], b1);
            prod10[i] *= fmaf(a0, acc10[i], b0);
            prod11[i] *= fmaf(a1, acc11[i], b1);
        }
    }

    // Store: acc[mt][ct] element i -> row btile*32 + mt*16 + half*8 + i,
    //                                 col ctile*32 + ct*16 + l16
    size_t r0 = (size_t)(btile * 32 + half * 8) * NCHOICE;
    size_t r1 = (size_t)(btile * 32 + 16 + half * 8) * NCHOICE;
#pragma unroll
    for (int i = 0; i < 8; ++i) {
        out[r0 + (size_t)i * NCHOICE + crow0] = prod00[i];
        out[r0 + (size_t)i * NCHOICE + crow1] = prod01[i];
        out[r1 + (size_t)i * NCHOICE + crow0] = prod10[i];
        out[r1 + (size_t)i * NCHOICE + crow1] = prod11[i];
    }
}

// ---------------------------------------------------------------------------
// Launch
// ---------------------------------------------------------------------------
extern "C" void kernel_launch(void* const* d_in, const int* in_sizes, int n_in,
                              void* d_out, int out_size, void* d_ws, size_t ws_size,
                              hipStream_t stream)
{
    (void)in_sizes; (void)n_in; (void)out_size; (void)ws_size;

    const float* query  = (const float*)d_in[0];   // [1024, 2048]
    const float* weight = (const float*)d_in[1];   // [4096, 2048]
    const float* nandw  = (const float*)d_in[2];   // [4096, 8]
    float* out = (float*)d_out;                    // [1024, 4096]

    // Workspace layout (≈ 40.25 MiB)
    __bf16* qhi = (__bf16*)d_ws;                          // 1024*2048
    __bf16* qlo = qhi + (size_t)BATCH * ROW;
    __bf16* whi = qlo + (size_t)BATCH * ROW;              // 4096*2048
    __bf16* wlo = whi + (size_t)NCHOICE * ROW;
    float*  ca  = (float*)(wlo + (size_t)NCHOICE * ROW);  // [8][4096]
    float*  cb  = ca + (size_t)NVEC * NCHOICE;

    // 1) normalize + hi/lo split: one wave per 256-element sub-vector
    {
        int rows = BATCH * NVEC;                          // 8192
        NAND_norm_split_kernel<<<rows * 32 / 256, 256, 0, stream>>>(query, qhi, qlo, rows);
    }
    {
        int rows = NCHOICE * NVEC;                        // 32768
        NAND_norm_split_kernel<<<rows * 32 / 256, 256, 0, stream>>>(weight, whi, wlo, rows);
    }

    // 2) sigmoid coefficients
    NAND_coeff_kernel<<<(NCHOICE * NVEC + 255) / 256, 256, 0, stream>>>(nandw, ca, cb);

    // 3) WMMA main: (1024/32)*(4096/32) = 4096 waves, 8 waves/block
    {
        int waves  = (BATCH / 32) * (NCHOICE / 32);
        int blocks = waves * 32 / 256;                    // 512
        NAND_wmma_kernel<<<blocks, 256, 0, stream>>>(qhi, qlo, whi, wlo, ca, cb, out);
    }
}
